// CostVolume_23304492548691
// MI455X (gfx1250) — compile-verified, matrix operand order checked by reading, not run on hardware
//
#include <hip/hip_runtime.h>
#include <hip/hip_bf16.h>
#include <stdint.h>

// ---------------------------------------------------------------------------
// Types
// ---------------------------------------------------------------------------
typedef _Float16 half_t;
typedef __attribute__((ext_vector_type(16))) _Float16 v16h;
typedef __attribute__((ext_vector_type(8)))  _Float16 h8;
typedef __attribute__((ext_vector_type(8)))  float    v8f;
typedef __attribute__((ext_vector_type(4)))  unsigned int v4u;
typedef __attribute__((ext_vector_type(8)))  int      v8i;
typedef __attribute__((ext_vector_type(4)))  int      v4i;

static constexpr int  Bsz = 2;
static constexpr int  Npt = 8192;
static constexpr int  Cch = 64;
static constexpr int  NSs = 8;
static constexpr int  NSQ = 32;
static constexpr long Pq  = (long)Bsz * Npt * NSQ;   // 524288 cross positions
static constexpr long Ps  = (long)Bsz * Npt * NSs;   // 131072 self positions
#define BN_EPS 1e-5f

#if defined(__gfx1250__) && __has_builtin(__builtin_amdgcn_tensor_load_to_lds)
#define USE_TDM 1
#else
#define USE_TDM 0
#endif

// ---------------------------------------------------------------------------
// Weight f32 -> f16 with zero K-padding.  W is [K][N] row-major.
// ---------------------------------------------------------------------------
__global__ void wpad_kernel(const float* __restrict__ W, half_t* __restrict__ O,
                            int K, int Nn, int Kpad)
{
    long e = (long)blockIdx.x * blockDim.x + threadIdx.x;
    if (e >= (long)Kpad * Nn) return;
    int k = (int)(e / Nn), n = (int)(e % Nn);
    O[e] = (half_t)(k < K ? W[(long)k * Nn + n] : 0.0f);
}

// ---------------------------------------------------------------------------
// Brute-force KNN (squared L2), register top-K, LDS-tiled source points.
// grid = (Npt/256, B); one query per thread.
// ---------------------------------------------------------------------------
template<int K>
__global__ __launch_bounds__(256) void knn_kernel(const float* __restrict__ q,
                                                  const float* __restrict__ s,
                                                  int* __restrict__ idx)
{
    __shared__ float sx[256], sy[256], sz[256];
    const int b = blockIdx.y;
    const float* qb = q + (long)b * Npt * 3;
    const float* sb = s + (long)b * Npt * 3;
    const int qi = blockIdx.x * 256 + threadIdx.x;
    const float qx = qb[qi * 3 + 0], qy = qb[qi * 3 + 1], qz = qb[qi * 3 + 2];

    float bd[K]; int bi[K];
#pragma unroll
    for (int j = 0; j < K; ++j) { bd[j] = 3.0e38f; bi[j] = 0; }
    float wd = 3.0e38f; int wsl = 0;

    for (int tile = 0; tile < Npt; tile += 256) {
        __syncthreads();
        const int j = tile + threadIdx.x;
        sx[threadIdx.x] = sb[j * 3 + 0];
        sy[threadIdx.x] = sb[j * 3 + 1];
        sz[threadIdx.x] = sb[j * 3 + 2];
        if (tile + 256 < Npt) __builtin_prefetch(sb + (tile + 256) * 3, 0, 1);
        __syncthreads();
#pragma unroll 4
        for (int u = 0; u < 256; ++u) {
            const float dx = sx[u] - qx, dy = sy[u] - qy, dz = sz[u] - qz;
            const float d = dx * dx + dy * dy + dz * dz;
            if (d < wd) {
                const int si = tile + u;
#pragma unroll
                for (int j2 = 0; j2 < K; ++j2)
                    if (j2 == wsl) { bd[j2] = d; bi[j2] = si; }
                float nw = -3.0e38f; int nsl = 0;
#pragma unroll
                for (int j2 = 0; j2 < K; ++j2)
                    if (bd[j2] > nw) { nw = bd[j2]; nsl = j2; }
                wd = nw; wsl = nsl;
            }
        }
    }
    const long base = ((long)b * Npt + qi) * K;
#pragma unroll
    for (int j = 0; j < K; ++j) idx[base + j] = bi[j];
}

// ---------------------------------------------------------------------------
// Feature builders (materialize f16 activation matrices)
// ---------------------------------------------------------------------------
__global__ void build_x1_kernel(const float* __restrict__ wxyz, const float* __restrict__ wpts,
                                const float* __restrict__ f2xyz, const float* __restrict__ f2pts,
                                const int* __restrict__ idxq, half_t* __restrict__ X1)
{
    long e = (long)blockIdx.x * blockDim.x + threadIdx.x;
    if (e >= Pq * 160) return;
    const int col = (int)(e % 160);
    const long r  = e / 160;
    const int b   = (int)(r / ((long)Npt * NSQ));
    const int rem = (int)(r % ((long)Npt * NSQ));
    const int n   = rem / NSQ;
    const int j   = idxq[r];
    const float* pw = wxyz  + ((long)b * Npt + n) * 3;
    const float* pf = f2xyz + ((long)b * Npt + j) * 3;
    float v = 0.0f;
    if (col < 3)        v = pw[col];
    else if (col < 6)   v = pf[col - 3];
    else if (col < 9)   v = pf[col - 6] - pw[col - 6];
    else if (col == 9) {
        float dx = pf[0] - pw[0], dy = pf[1] - pw[1], dz = pf[2] - pw[2];
        v = sqrtf(dx * dx + dy * dy + dz * dz + 1e-20f);
    }
    else if (col < 74)  v = wpts [((long)b * Npt + n) * Cch + (col - 10)];
    else if (col < 138) v = f2pts[((long)b * Npt + j) * Cch + (col - 74)];
    X1[e] = (half_t)v;
}

__global__ void build_x2a_kernel(const float* __restrict__ wxyz, const int* __restrict__ idxs,
                                 half_t* __restrict__ X2a)
{
    long e = (long)blockIdx.x * blockDim.x + threadIdx.x;
    if (e >= Ps * 32) return;
    const int col = (int)(e % 32);
    const long r  = e / 32;
    const int b   = (int)(r / ((long)Npt * NSs));
    const int rem = (int)(r % ((long)Npt * NSs));
    const int n   = rem / NSs;
    const int j   = idxs[r];
    const float* pw = wxyz + ((long)b * Npt + n) * 3;
    const float* pg = wxyz + ((long)b * Npt + j) * 3;
    float v = 0.0f;
    if (col < 3)        v = pw[col];
    else if (col < 6)   v = pg[col - 3];
    else if (col < 9)   v = pg[col - 6] - pw[col - 6];
    else if (col == 9) {
        float dx = pg[0] - pw[0], dy = pg[1] - pw[1], dz = pg[2] - pw[2];
        v = sqrtf(dx * dx + dy * dy + dz * dz + 1e-20f);
    }
    X2a[e] = (half_t)v;
}

__global__ void build_pnp_kernel(const float* __restrict__ wpts, const float* __restrict__ pifeat,
                                 const int* __restrict__ idxs,
                                 half_t* __restrict__ PNEW, half_t* __restrict__ PGRP)
{
    long e = (long)blockIdx.x * blockDim.x + threadIdx.x;
    if (e >= Ps * Cch) return;
    const int c  = (int)(e % Cch);
    const long r = e / Cch;
    const int b   = (int)(r / ((long)Npt * NSs));
    const int rem = (int)(r % ((long)Npt * NSs));
    const int n   = rem / NSs;
    const int j   = idxs[r];
    PNEW[e] = (half_t)wpts  [((long)b * Npt + n) * Cch + c];
    PGRP[e] = (half_t)pifeat[((long)b * Npt + j) * Cch + c];
}

// ---------------------------------------------------------------------------
// WMMA GEMM:  Y[M x NOUT] = A[M x KPAD] * W[KPAD x NOUT]   (f16 in, f32 acc)
// Weight tile is DMA'd into LDS by the Tensor Data Mover (one transfer per
// block, issued by wave 0, tracked by TENSORcnt), then swizzled LDS->LDS into
// WMMA B-fragment order so each fragment is a single 32B ds read.
// ---------------------------------------------------------------------------
template<int KSTEPS, int NTILES>
__global__ __launch_bounds__(256) void gemm_wmma(
    const half_t* __restrict__ A0, int lda0,
    const half_t* __restrict__ A1, int lda1,
    const half_t* __restrict__ A2, int lda2,
    int segw,
    const half_t* __restrict__ Wp,     // [KSTEPS*32][NTILES*16]
    half_t* __restrict__ Yout)
{
    constexpr int NOUT = NTILES * 16;
    constexpr int KPAD = KSTEPS * 32;
    __shared__ __align__(32) half_t bfr[NTILES * KSTEPS * 32 * 16];
    const int tid = threadIdx.x;

#if USE_TDM
    // ---- TDM: DMA the whole weight tile (KPAD x NOUT f16) into LDS ----
    __shared__ __align__(16) half_t wlds[KPAD * NOUT];
    if (tid < 32) {
        const unsigned lds_off = (unsigned)(uintptr_t)(void*)wlds;
        const unsigned long long ga = (unsigned long long)(uintptr_t)(const void*)Wp;
        v4u g0;
        g0[0] = 1u;                                   // count=1 (valid descriptor)
        g0[1] = lds_off;                              // lds_addr
        g0[2] = (unsigned)(ga & 0xFFFFFFFFull);       // global_addr[31:0]
        g0[3] = (unsigned)((ga >> 32) & 0x01FFFFFFull) | (2u << 30);  // addr[56:32] | type=2
        v8i g1;
        g1[0] = 1 << 16;                              // data_size = 2 bytes
        g1[1] = NOUT << 16;                           // tensor_dim0[15:0]
        g1[2] = KPAD << 16;                           // tensor_dim0 hi=0 | tensor_dim1[15:0]
        g1[3] = NOUT << 16;                           // tensor_dim1 hi=0 | tile_dim0
        g1[4] = KPAD;                                 // tile_dim1 | tile_dim2=0
        g1[5] = NOUT;                                 // tensor_dim0_stride[31:0]
        g1[6] = 0;                                    // stride0 hi | stride1 lo
        g1[7] = 0;
        const v4i gz = {0, 0, 0, 0};
#if __clang_major__ >= 23
        const v8i gz8 = {0, 0, 0, 0, 0, 0, 0, 0};
        __builtin_amdgcn_tensor_load_to_lds(g0, g1, gz, gz, gz8, 0);
#else
        __builtin_amdgcn_tensor_load_to_lds(g0, g1, gz, gz, 0);
#endif
        __builtin_amdgcn_s_wait_tensorcnt(0);
    }
    __syncthreads();
#endif

    // ---- swizzle W into B-fragment layout ----
    constexpr int TOT = NTILES * KSTEPS * 32 * 16;
    for (int e = tid; e < TOT; e += 256) {
        const int i = e & 15;
        const int f = e >> 4;
        const int L = f & 31;
        const int s = (f >> 5) % KSTEPS;
        const int t = f / (32 * KSTEPS);
        const int n   = t * 16 + (L & 15);
        const int klo = s * 32 + ((L & 16) >> 1);
        const int k   = klo + ((i < 8) ? i : (8 + i));
#if USE_TDM
        bfr[e] = wlds[k * NOUT + n];
#else
        bfr[e] = Wp[(long)k * NOUT + n];
#endif
    }
    __syncthreads();

    const int lane = tid & 31;
    const int wave = tid >> 5;
    const long m0  = (long)blockIdx.x * 128 + wave * 16;
    const long r   = m0 + (lane & 15);
    const int klo  = (lane & 16) >> 1;     // 0 or 8

    // ---- load all A fragments for this wave's 16 rows ----
    v16h afr[KSTEPS];
#pragma unroll
    for (int s = 0; s < KSTEPS; ++s) {
        const int ks  = s * 32;
        const int seg = ks / segw;
        const half_t* Ab; int lda;
        if (seg == 0)      { Ab = A0; lda = lda0; }
        else if (seg == 1) { Ab = A1; lda = lda1; }
        else               { Ab = A2; lda = lda2; }
        const int kin = ks - seg * segw + klo;
        const half_t* p = Ab + r * lda + kin;
        const h8 lo = *(const h8*)(p);
        const h8 hi = *(const h8*)(p + 16);
        v16h a;
#pragma unroll
        for (int i = 0; i < 8; ++i) { a[i] = lo[i]; a[i + 8] = hi[i]; }
        afr[s] = a;
    }

    // ---- NTILES output tiles, KSTEPS wmma each ----
#pragma unroll
    for (int t = 0; t < NTILES; ++t) {
        v8f acc = {};
#pragma unroll
        for (int s = 0; s < KSTEPS; ++s) {
            const v16h b = *(const v16h*)&bfr[(((t * KSTEPS) + s) * 32 + lane) * 16];
            acc = __builtin_amdgcn_wmma_f32_16x16x32_f16(
                false, afr[s], false, b, (short)0, acc, false, false);
        }
        const int  nb = t * 16 + (lane & 15);
        const long rb = m0 + ((lane >> 4) << 3);
#pragma unroll
        for (int v = 0; v < 8; ++v)
            Yout[(rb + v) * NOUT + nb] = (half_t)acc[v];
    }
}

// ---------------------------------------------------------------------------
// Training-mode BatchNorm (+ReLU) on f16 buffer [M x Cc] — 3 phases
// ---------------------------------------------------------------------------
__global__ __launch_bounds__(256) void bn_stats_kernel(const half_t* __restrict__ Y,
                                                       long M, int Cc,
                                                       float* __restrict__ partial)
{
    const int tid = threadIdx.x;
    const int c = tid % Cc;
    const int g = tid / Cc;
    const int G = 256 / Cc;
    const long rows = (M + gridDim.x - 1) / gridDim.x;
    const long r0 = (long)blockIdx.x * rows;
    long r1 = r0 + rows; if (r1 > M) r1 = M;
    float s = 0.0f, q = 0.0f;
    for (long r = r0 + g; r < r1; r += G) {
        const float v = (float)Y[r * Cc + c];
        s += v; q += v * v;
    }
    __shared__ float sh[512];
    sh[tid] = s; sh[256 + tid] = q;
    __syncthreads();
    if (g == 0) {
        for (int gg = 1; gg < G; ++gg) { s += sh[gg * Cc + c]; q += sh[256 + gg * Cc + c]; }
        partial[((long)blockIdx.x * Cc + c) * 2 + 0] = s;
        partial[((long)blockIdx.x * Cc + c) * 2 + 1] = q;
    }
}

__global__ void bn_finalize_kernel(const float* __restrict__ partial,
                                   const float* __restrict__ gamma,
                                   const float* __restrict__ beta,
                                   long M, int Cc, float* __restrict__ bnp)
{
    const int c = threadIdx.x;
    if (c >= Cc) return;
    float s = 0.0f, q = 0.0f;
    for (int b2 = 0; b2 < 256; ++b2) {
        s += partial[((long)b2 * Cc + c) * 2 + 0];
        q += partial[((long)b2 * Cc + c) * 2 + 1];
    }
    const float mean = s / (float)M;
    const float var  = q / (float)M - mean * mean;
    const float sc   = gamma[c] * rsqrtf(var + BN_EPS);
    bnp[c * 2 + 0] = sc;
    bnp[c * 2 + 1] = beta[c] - mean * sc;
}

__global__ void bn_apply_kernel(half_t* __restrict__ Y, long total, int Cc,
                                const float* __restrict__ bnp)
{
    long e = (long)blockIdx.x * blockDim.x + threadIdx.x;
    if (e >= total) return;
    const int c = (int)(e % Cc);
    const float v = (float)Y[e] * bnp[c * 2] + bnp[c * 2 + 1];
    Y[e] = (half_t)(v > 0.0f ? v : 0.0f);
}

// ---------------------------------------------------------------------------
// Softmax over K axis + weighted sum:  out[b,n,c] = sum_k softmax_k(Z)*F
// ---------------------------------------------------------------------------
template<int K>
__global__ void softmax_ws_kernel(const half_t* __restrict__ Z, const half_t* __restrict__ F,
                                  float* __restrict__ out, long total)
{
    long t = (long)blockIdx.x * blockDim.x + threadIdx.x;
    if (t >= total) return;
    const int  c  = (int)(t % Cch);
    const long bn = t / Cch;
    const long base = bn * (long)K * Cch + c;
    float m = -3.0e38f;
#pragma unroll
    for (int k = 0; k < K; ++k) {
        const float z = (float)Z[base + (long)k * Cch];
        m = fmaxf(m, z);
    }
    float den = 0.0f, num = 0.0f;
#pragma unroll
    for (int k = 0; k < K; ++k) {
        const float z = (float)Z[base + (long)k * Cch];
        const float e = __expf(z - m);
        den += e;
        num += e * (float)F[base + (long)k * Cch];
    }
    out[t] = num / den;
}

// ---------------------------------------------------------------------------
// Host orchestration
// ---------------------------------------------------------------------------
extern "C" void kernel_launch(void* const* d_in, const int* in_sizes, int n_in,
                              void* d_out, int out_size, void* d_ws, size_t ws_size,
                              hipStream_t stream)
{
    (void)in_sizes; (void)n_in; (void)out_size; (void)ws_size;

    const float* wxyz  = (const float*)d_in[0];
    const float* wpts  = (const float*)d_in[1];
    const float* f2xyz = (const float*)d_in[2];
    const float* f2pts = (const float*)d_in[3];

    const float* W_m1[3] = { (const float*)d_in[4],  (const float*)d_in[8],  (const float*)d_in[12] };
    const float* G_m1[3] = { (const float*)d_in[6],  (const float*)d_in[10], (const float*)d_in[14] };
    const float* E_m1[3] = { (const float*)d_in[7],  (const float*)d_in[11], (const float*)d_in[15] };
    const float* W_pi = (const float*)d_in[16]; const float* G_pi = (const float*)d_in[18]; const float* E_pi = (const float*)d_in[19];
    const float* W_pc = (const float*)d_in[20]; const float* G_pc = (const float*)d_in[22]; const float* E_pc = (const float*)d_in[23];
    const float* W_m2[2] = { (const float*)d_in[24], (const float*)d_in[28] };
    const float* G_m2[2] = { (const float*)d_in[26], (const float*)d_in[30] };
    const float* E_m2[2] = { (const float*)d_in[27], (const float*)d_in[31] };
    const float* W_mb[2] = { (const float*)d_in[32], (const float*)d_in[36] };
    const float* G_mb[2] = { (const float*)d_in[34], (const float*)d_in[38] };
    const float* E_mb[2] = { (const float*)d_in[35], (const float*)d_in[39] };

    // ---- workspace bump arena ----
    char* base = (char*)d_ws;
    size_t off = 0;
    auto alloc = [&](size_t bytes) -> char* {
        char* p = base + off;
        off += (bytes + 255) & ~size_t(255);
        return p;
    };

    half_t* Wp0  = (half_t*)alloc(160 * 128 * 2);
    half_t* Wp1  = (half_t*)alloc(128 *  64 * 2);
    half_t* Wp2  = (half_t*)alloc( 64 *  64 * 2);
    half_t* Wppi = (half_t*)alloc( 32 *  64 * 2);
    half_t* Wppc = (half_t*)alloc( 32 *  64 * 2);
    half_t* Wpm0 = (half_t*)alloc(128 * 128 * 2);
    half_t* Wpm1 = (half_t*)alloc(128 *  64 * 2);
    half_t* Wpb0 = (half_t*)alloc(192 * 128 * 2);
    half_t* Wpb1 = (half_t*)alloc(128 *  64 * 2);
    float*  partial = (float*)alloc(256 * 128 * 2 * 4);
    float*  bnp     = (float*)alloc(128 * 2 * 4);
    int*    idxq    = (int*)alloc(Pq * 4);
    int*    idxs    = (int*)alloc(Ps * 4);
    half_t* X1 = (half_t*)alloc(Pq * 160 * 2);     // also hosts Z1 (Pq x 128)
    half_t* Y1 = (half_t*)alloc(Pq * 128 * 2);     // also hosts Z2 (Pq x 64)
    half_t* Y2 = (half_t*)alloc(Pq *  64 * 2);     // also hosts U1, U2
    half_t* Y3 = (half_t*)alloc(Pq *  64 * 2);     // feat (kept live)
    half_t* E  = (half_t*)alloc(Pq *  64 * 2);     // pi_enc out; later hosts self-path bufs
    float*  pifeat = (float*)alloc((long)Bsz * Npt * Cch * 4);

    half_t* Z1  = X1;
    half_t* Z2  = Y1;
    half_t* U1  = Y2;
    half_t* U2  = Y2 + Ps * 128;
    half_t* X2a  = E;
    half_t* PNEW = E + Ps * 32;
    half_t* PGRP = PNEW + Ps * 64;
    half_t* EPC  = PGRP + Ps * 64;

    auto gl = [](long n) { return (unsigned)((n + 255) / 256); };

    // ---- 1) weight prep ----
    wpad_kernel<<<gl(160 * 128), 256, 0, stream>>>(W_m1[0], Wp0, 138, 128, 160);
    wpad_kernel<<<gl(128 *  64), 256, 0, stream>>>(W_m1[1], Wp1, 128,  64, 128);
    wpad_kernel<<<gl( 64 *  64), 256, 0, stream>>>(W_m1[2], Wp2,  64,  64,  64);
    wpad_kernel<<<gl( 32 *  64), 256, 0, stream>>>(W_pi,   Wppi,  10,  64,  32);
    wpad_kernel<<<gl( 32 *  64), 256, 0, stream>>>(W_pc,   Wppc,  10,  64,  32);
    wpad_kernel<<<gl(128 * 128), 256, 0, stream>>>(W_m2[0], Wpm0, 128, 128, 128);
    wpad_kernel<<<gl(128 *  64), 256, 0, stream>>>(W_m2[1], Wpm1, 128,  64, 128);
    wpad_kernel<<<gl(192 * 128), 256, 0, stream>>>(W_mb[0], Wpb0, 192, 128, 192);
    wpad_kernel<<<gl(128 *  64), 256, 0, stream>>>(W_mb[1], Wpb1, 128,  64, 128);

    // ---- 2) KNN ----
    knn_kernel<NSQ><<<dim3(Npt / 256, Bsz), 256, 0, stream>>>(wxyz, f2xyz, idxq);
    knn_kernel<NSs><<<dim3(Npt / 256, Bsz), 256, 0, stream>>>(wxyz, wxyz,  idxs);

    auto BN = [&](half_t* buf, long M, int Cc, const float* g, const float* be) {
        bn_stats_kernel<<<256, 256, 0, stream>>>(buf, M, Cc, partial);
        bn_finalize_kernel<<<1, 128, 0, stream>>>(partial, g, be, M, Cc, bnp);
        bn_apply_kernel<<<gl(M * Cc), 256, 0, stream>>>(buf, M * Cc, Cc, bnp);
    };

    // ---- 3) cross path ----
    build_x1_kernel<<<gl(Pq * 160), 256, 0, stream>>>(wxyz, wpts, f2xyz, f2pts, idxq, X1);

    gemm_wmma<5, 8><<<Pq / 128, 256, 0, stream>>>(X1, 160, nullptr, 0, nullptr, 0, 160, Wp0, Y1);
    BN(Y1, Pq, 128, G_m1[0], E_m1[0]);
    gemm_wmma<4, 4><<<Pq / 128, 256, 0, stream>>>(Y1, 128, nullptr, 0, nullptr, 0, 128, Wp1, Y2);
    BN(Y2, Pq, 64, G_m1[1], E_m1[1]);
    gemm_wmma<2, 4><<<Pq / 128, 256, 0, stream>>>(Y2, 64, nullptr, 0, nullptr, 0, 64, Wp2, Y3);
    BN(Y3, Pq, 64, G_m1[2], E_m1[2]);

    // pi_enc reads X1 cols 0..31 (W rows 10..31 are zero -> padding irrelevant)
    gemm_wmma<1, 4><<<Pq / 128, 256, 0, stream>>>(X1, 160, nullptr, 0, nullptr, 0, 32, Wppi, E);
    BN(E, Pq, 64, G_pi, E_pi);

    gemm_wmma<4, 8><<<Pq / 128, 256, 0, stream>>>(E, 64, Y3, 64, nullptr, 0, 64, Wpm0, Z1);
    BN(Z1, Pq, 128, G_m2[0], E_m2[0]);
    gemm_wmma<4, 4><<<Pq / 128, 256, 0, stream>>>(Z1, 128, nullptr, 0, nullptr, 0, 128, Wpm1, Z2);
    BN(Z2, Pq, 64, G_m2[1], E_m2[1]);

    softmax_ws_kernel<NSQ><<<gl((long)Bsz * Npt * Cch), 256, 0, stream>>>(
        Z2, Y3, pifeat, (long)Bsz * Npt * Cch);

    // ---- 4) self path ----
    build_x2a_kernel<<<gl(Ps * 32), 256, 0, stream>>>(wxyz, idxs, X2a);
    gemm_wmma<1, 4><<<Ps / 128, 256, 0, stream>>>(X2a, 32, nullptr, 0, nullptr, 0, 32, Wppc, EPC);
    BN(EPC, Ps, 64, G_pc, E_pc);

    build_pnp_kernel<<<gl(Ps * Cch), 256, 0, stream>>>(wpts, pifeat, idxs, PNEW, PGRP);

    gemm_wmma<6, 8><<<Ps / 128, 256, 0, stream>>>(EPC, 64, PNEW, 64, PGRP, 64, 64, Wpb0, U1);
    BN(U1, Ps, 128, G_mb[0], E_mb[0]);
    gemm_wmma<4, 4><<<Ps / 128, 256, 0, stream>>>(U1, 128, nullptr, 0, nullptr, 0, 128, Wpb1, U2);
    BN(U2, Ps, 64, G_mb[1], E_mb[1]);

    softmax_ws_kernel<NSs><<<gl((long)Bsz * Npt * Cch), 256, 0, stream>>>(
        U2, PGRP, (float*)d_out, (long)Bsz * Npt * Cch);
}
